// MultiModalFusionGNN_28278064677000
// MI455X (gfx1250) — compile-verified, matrix operand order checked by reading, not run on hardware
//
#include <hip/hip_runtime.h>
#include <hip/hip_bf16.h>

typedef _Float16 half_t;
typedef __attribute__((ext_vector_type(16))) _Float16 v16h;
typedef __attribute__((ext_vector_type(8)))  _Float16 v8h;
typedef __attribute__((ext_vector_type(8)))  float    v8f;

// ---------------------------------------------------------------------------
// Stage 1: dense f16 weights in workspace; block-diagonal [32x192] encoder
// weight (K padded 30->32) and concatenated encoder bias [192].
// ---------------------------------------------------------------------------
__global__ void prep_dense(const float* __restrict__ wf, const float* __restrict__ ww,
                           const float* __restrict__ wt,
                           const float* __restrict__ bf, const float* __restrict__ bw,
                           const float* __restrict__ bt,
                           const float* __restrict__ g0, const float* __restrict__ g1,
                           const float* __restrict__ g2, const float* __restrict__ mw1,
                           half_t* __restrict__ wenc, half_t* __restrict__ w0h,
                           half_t* __restrict__ w1h, half_t* __restrict__ w2h,
                           half_t* __restrict__ wm1, float* __restrict__ benc)
{
    int i = blockIdx.x * blockDim.x + threadIdx.x;
    if (i < 32 * 192) {                      // block-diagonal encoder weight
        int k = i / 192, c = i % 192;
        float v = 0.0f;
        if (c < 64)       { if (k < 8)              v = wf[k * 64 + c];              }
        else if (c < 128) { if (k >= 8 && k < 20)   v = ww[(k - 8) * 64 + (c - 64)]; }
        else              { if (k >= 20 && k < 30)  v = wt[(k - 20) * 64 + (c - 128)]; }
        wenc[i] = (half_t)v;
    }
    if (i < 192 * 64) w0h[i] = (half_t)g0[i];
    if (i < 64 * 64)  { w1h[i] = (half_t)g1[i]; w2h[i] = (half_t)g2[i]; }
    if (i < 64 * 32)  wm1[i] = (half_t)mw1[i];
    if (i < 192)      benc[i] = (i < 64) ? bf[i] : (i < 128) ? bw[i - 64] : bt[i - 128];
}

// ---------------------------------------------------------------------------
// Stage 2: swizzle dense W[K][Nc] into the per-lane WMMA B-fragment layout so
// each lane reads one contiguous 32B v16h per (n-tile, k-chunk).
// Packed index p = ((tn*(K/32)+c)*32 + lane)*16 + j, where fragment element j
// maps to k = c*32 + ((j>>1 <4) ? 2*(j>>1) : 16+2*((j>>1)-4)) + 8*(lane>>4) + (j&1),
// col = tn*16 + (lane&15)   (CDNA5 ISA 7.12.2 16-bit operand layout).
// ---------------------------------------------------------------------------
__global__ void pack_btiles(const half_t* __restrict__ src, half_t* __restrict__ dst,
                            int K, int Nc)
{
    int p = blockIdx.x * blockDim.x + threadIdx.x;
    if (p >= K * Nc) return;
    int j    = p & 15;
    int lane = (p >> 4) & 31;
    int rest = p >> 9;                 // tn*(K/32) + c
    int chunks = K >> 5;
    int c  = rest % chunks;
    int tn = rest / chunks;
    int i16 = lane & 15, hh = lane >> 4;
    int v = j >> 1, par = j & 1;
    int kb = ((v < 4) ? 2 * v : 16 + 2 * (v - 4)) + 8 * hh + par;
    int k  = c * 32 + kb;
    int col = tn * 16 + i16;
    dst[p] = src[(size_t)k * Nc + col];
}

// Pack raw node features into the K=32 padded f16 input of the encoder GEMM.
__global__ void pack_input(const float* __restrict__ xf, const float* __restrict__ xw,
                           const float* __restrict__ xt, half_t* __restrict__ xin, int Nn)
{
    long long idx = (long long)blockIdx.x * blockDim.x + threadIdx.x;
    if (idx >= (long long)Nn * 32) return;
    int i = (int)(idx >> 5), c = (int)(idx & 31);
    float v = 0.0f;
    if (c < 8)        v = xf[(size_t)i * 8  + c];
    else if (c < 20)  v = xw[(size_t)i * 12 + (c - 8)];
    else if (c < 30)  v = xt[(size_t)i * 10 + (c - 20)];
    xin[idx] = (half_t)v;
}

// ---------------------------------------------------------------------------
// WMMA GEMM: C[M,Nc] = A[M,K](f16) @ W(packed f16), f32 accumulate.
// One wave32 per 16x(16*NT) output strip: A fragment loaded once per K-chunk
// (two contiguous v8h = global_load_b128 each) and reused for NT v_wmma
// issues; B fragments are single contiguous v16h loads from the packed layout.
// Compile-time BIAS/RELU keep the epilogue branch-free.
// ---------------------------------------------------------------------------
template <typename OutT, bool BIAS, bool RELU, int NT>
__global__ __launch_bounds__(256) void wmma_gemm(const half_t* __restrict__ A,
                                                 const half_t* __restrict__ Wp,
                                                 OutT* __restrict__ C,
                                                 int M, int K, int Nc,
                                                 const float* __restrict__ bias)
{
    int tilesM = M >> 4;
    int groups = (Nc >> 4) / NT;
    int tile   = blockIdx.x * 8 + (threadIdx.x >> 5);   // 8 waves / block
    if (tile >= tilesM * groups) return;                // uniform per wave
    int tm = tile % tilesM, tg = tile / tilesM;
    int lane = threadIdx.x & 31;
    int i16 = lane & 15, hh = lane >> 4;
    const half_t* Arow = A + (size_t)(tm * 16 + i16) * K;
    int chunks = K >> 5;

    v8f acc[NT] = {};
    for (int c = 0; c < chunks; ++c) {
        union { v16h v; v8h h[2]; } a;
        a.h[0] = *(const v8h*)(Arow + c * 32 + 8 * hh);        // K = c*32 + 8h .. +7
        a.h[1] = *(const v8h*)(Arow + c * 32 + 16 + 8 * hh);   // K = c*32+16+8h .. +7
#pragma unroll
        for (int t = 0; t < NT; ++t) {
            const v16h b = *(const v16h*)(Wp +
                ((((size_t)(tg * NT + t) * chunks + c) << 5) + lane) * 16);
            acc[t] = __builtin_amdgcn_wmma_f32_16x16x32_f16(
                false, a.v, false, b, (short)0, acc[t], false, false);
        }
    }
#pragma unroll
    for (int t = 0; t < NT; ++t) {
        int col = (tg * NT + t) * 16 + i16;
        float bv = BIAS ? bias[col] : 0.0f;
#pragma unroll
        for (int r = 0; r < 8; ++r) {                  // C VGPR r -> M = r + 8*half
            float v = acc[t][r] + bv;
            if (RELU) v = v > 0.0f ? v : 0.0f;
            C[(size_t)(tm * 16 + r + 8 * hh) * Nc + col] = (OutT)v;
        }
    }
}

// ---------------------------------------------------------------------------
// Graph kernels
// ---------------------------------------------------------------------------
__global__ void zero_f32(float* __restrict__ p, long long n)
{
    long long i = (long long)blockIdx.x * blockDim.x + threadIdx.x;
    if (i < n) p[i] = 0.0f;
}

__global__ void compute_deg(const int* __restrict__ eidx, float* __restrict__ deg, int E)
{
    int e = blockIdx.x * blockDim.x + threadIdx.x;
    if (e < E) atomicAdd(&deg[eidx[E + e]], 1.0f);      // dst degrees
}

__global__ void finalize_dinv(float* __restrict__ d, int Nn)
{
    int i = blockIdx.x * blockDim.x + threadIdx.x;
    if (i < Nn) d[i] = rsqrtf(d[i] + 1.0f);             // +1 self-loop
}

// Feature-parallel edge scatter: 16 lanes per edge, 4 floats per lane.
// Coalesced float4 reads of h[src]; f32 atomics into agg[dst].
__global__ void scatter_edges(const int* __restrict__ eidx, const float* __restrict__ dinv,
                              const float* __restrict__ h, float* __restrict__ agg, int E)
{
    long long idx = (long long)blockIdx.x * blockDim.x + threadIdx.x;
    if (idx >= (long long)E * 16) return;
    int e    = (int)(idx >> 4);
    int part = (int)(idx & 15);
    int src = eidx[e], dst = eidx[E + e];
    float nrm = dinv[src] * dinv[dst];
    int f = part * 4;
    const float4 hv = *(const float4*)(h + (size_t)src * 64 + f);
    float* ad = agg + (size_t)dst * 64 + f;
    atomicAdd(ad + 0, hv.x * nrm);
    atomicAdd(ad + 1, hv.y * nrm);
    atomicAdd(ad + 2, hv.z * nrm);
    atomicAdd(ad + 3, hv.w * nrm);
}

// out = relu(agg + h*dinv^2 + b), emitted as f16 for the next WMMA GEMM.
__global__ void combine(const float* __restrict__ agg, const float* __restrict__ h,
                        const float* __restrict__ dinv, const float* __restrict__ b,
                        half_t* __restrict__ xout, int Nn)
{
    long long idx = (long long)blockIdx.x * blockDim.x + threadIdx.x;
    if (idx >= (long long)Nn * 64) return;
    int i = (int)(idx >> 6), f = (int)(idx & 63);
    float d = dinv[i];
    float v = agg[idx] + h[idx] * d * d + b[f];
    xout[idx] = (half_t)(v > 0.0f ? v : 0.0f);
}

// Final 32->1 projection (no ReLU on the output, per reference).
__global__ void final_out(const half_t* __restrict__ m1, const float* __restrict__ w2,
                          const float* __restrict__ b2, float* __restrict__ out, int Nn)
{
    int n = blockIdx.x * blockDim.x + threadIdx.x;
    if (n >= Nn) return;
    float acc = b2[0];
#pragma unroll
    for (int j = 0; j < 32; ++j) acc += (float)m1[(size_t)n * 32 + j] * w2[j];
    out[n] = acc;
}

// ---------------------------------------------------------------------------
// Launch
// ---------------------------------------------------------------------------
extern "C" void kernel_launch(void* const* d_in, const int* in_sizes, int n_in,
                              void* d_out, int out_size, void* d_ws, size_t ws_size,
                              hipStream_t stream)
{
    const float* x_fire = (const float*)d_in[0];
    const float* x_wea  = (const float*)d_in[1];
    const float* x_ter  = (const float*)d_in[2];
    const int*   eidx   = (const int*)d_in[3];
    const float* w_fire = (const float*)d_in[4];
    const float* b_fire = (const float*)d_in[5];
    const float* w_wea  = (const float*)d_in[6];
    const float* b_wea  = (const float*)d_in[7];
    const float* w_ter  = (const float*)d_in[8];
    const float* b_ter  = (const float*)d_in[9];
    const float* gcn_w0 = (const float*)d_in[10];
    const float* gcn_b0 = (const float*)d_in[11];
    const float* gcn_w1 = (const float*)d_in[12];
    const float* gcn_b1 = (const float*)d_in[13];
    const float* gcn_w2 = (const float*)d_in[14];
    const float* gcn_b2 = (const float*)d_in[15];
    const float* out_w1 = (const float*)d_in[16];
    const float* out_b1 = (const float*)d_in[17];
    const float* out_w2 = (const float*)d_in[18];
    const float* out_b2 = (const float*)d_in[19];
    float* out = (float*)d_out;

    const int Nn = in_sizes[0] / 8;      // 100000
    const int E  = in_sizes[3] / 2;      // 3200000

    // Workspace layout (256B aligned slices)
    char*  base = (char*)d_ws;
    size_t o = 0;
    auto alloc = [&](size_t bytes) -> void* {
        o = (o + 255) & ~(size_t)255;
        void* p = base + o;
        o += bytes;
        return p;
    };
    half_t* xin   = (half_t*)alloc((size_t)Nn * 32 * sizeof(half_t));   // encoder input
    half_t* x0    = (half_t*)alloc((size_t)Nn * 192 * sizeof(half_t));  // encoder output
    half_t* xcur  = (half_t*)alloc((size_t)Nn * 64 * sizeof(half_t));   // per-layer activations
    float*  hbuf  = (float*) alloc((size_t)Nn * 64 * sizeof(float));    // h = x @ W
    float*  agg   = (float*) alloc((size_t)Nn * 64 * sizeof(float));    // scatter accumulator
    float*  dinv  = (float*) alloc((size_t)Nn * sizeof(float));         // deg^-1/2
    half_t* wencD = (half_t*)alloc(32 * 192 * sizeof(half_t));          // dense staging
    half_t* w0D   = (half_t*)alloc(192 * 64 * sizeof(half_t));
    half_t* w1D   = (half_t*)alloc(64 * 64 * sizeof(half_t));
    half_t* w2D   = (half_t*)alloc(64 * 64 * sizeof(half_t));
    half_t* wm1D  = (half_t*)alloc(64 * 32 * sizeof(half_t));
    half_t* wencP = (half_t*)alloc(32 * 192 * sizeof(half_t));          // packed B tiles
    half_t* w0P   = (half_t*)alloc(192 * 64 * sizeof(half_t));
    half_t* w1P   = (half_t*)alloc(64 * 64 * sizeof(half_t));
    half_t* w2P   = (half_t*)alloc(64 * 64 * sizeof(half_t));
    half_t* wm1P  = (half_t*)alloc(64 * 32 * sizeof(half_t));
    float*  benc  = (float*) alloc(192 * sizeof(float));
    half_t* m1    = (half_t*)alloc((size_t)Nn * 32 * sizeof(half_t));   // MLP hidden
    (void)ws_size; (void)n_in; (void)out_size;

    const int BT = 256;
    auto blocks = [](long long n, int bt) { return (unsigned)((n + bt - 1) / bt); };

    // 1) weight prep (dense f16 -> packed B-fragment layout), input pack, degrees
    prep_dense<<<blocks(192 * 64, BT), BT, 0, stream>>>(
        w_fire, w_wea, w_ter, b_fire, b_wea, b_ter,
        gcn_w0, gcn_w1, gcn_w2, out_w1, wencD, w0D, w1D, w2D, wm1D, benc);
    pack_btiles<<<blocks(32 * 192, BT), BT, 0, stream>>>(wencD, wencP, 32, 192);
    pack_btiles<<<blocks(192 * 64, BT), BT, 0, stream>>>(w0D, w0P, 192, 64);
    pack_btiles<<<blocks(64 * 64, BT), BT, 0, stream>>>(w1D, w1P, 64, 64);
    pack_btiles<<<blocks(64 * 64, BT), BT, 0, stream>>>(w2D, w2P, 64, 64);
    pack_btiles<<<blocks(64 * 32, BT), BT, 0, stream>>>(wm1D, wm1P, 64, 32);
    pack_input<<<blocks((long long)Nn * 32, BT), BT, 0, stream>>>(x_fire, x_wea, x_ter, xin, Nn);
    zero_f32<<<blocks(Nn, BT), BT, 0, stream>>>(dinv, Nn);
    compute_deg<<<blocks(E, BT), BT, 0, stream>>>(eidx, dinv, E);
    finalize_dinv<<<blocks(Nn, BT), BT, 0, stream>>>(dinv, Nn);

    // 2) fused modality encoders: x0 = relu(xin @ wenc + benc)   [N,32]@[32,192]
    {
        long long waves = (long long)(Nn / 16) * ((192 / 16) / 4);
        wmma_gemm<half_t, true, true, 4><<<blocks(waves, 8), BT, 0, stream>>>(
            xin, wencP, x0, Nn, 32, 192, benc);
    }

    // 3) three GCN layers
    const half_t* Wl[3] = { w0P, w1P, w2P };
    const float*  bl[3] = { gcn_b0, gcn_b1, gcn_b2 };
    const int     Kl[3] = { 192, 64, 64 };
    for (int l = 0; l < 3; ++l) {
        const half_t* Ain = (l == 0) ? x0 : xcur;
        long long waves = (long long)(Nn / 16) * ((64 / 16) / 4);
        wmma_gemm<float, false, false, 4><<<blocks(waves, 8), BT, 0, stream>>>(
            Ain, Wl[l], hbuf, Nn, Kl[l], 64, nullptr);
        zero_f32<<<blocks((long long)Nn * 64, BT), BT, 0, stream>>>(agg, (long long)Nn * 64);
        scatter_edges<<<blocks((long long)E * 16, BT), BT, 0, stream>>>(eidx, dinv, hbuf, agg, E);
        combine<<<blocks((long long)Nn * 64, BT), BT, 0, stream>>>(agg, hbuf, dinv, bl[l], xcur, Nn);
    }

    // 4) output MLP: m1 = relu(x @ out_w1 + out_b1), out = m1 @ out_w2 + out_b2
    {
        long long waves = (long long)(Nn / 16) * ((32 / 16) / 2);
        wmma_gemm<half_t, true, true, 2><<<blocks(waves, 8), BT, 0, stream>>>(
            xcur, wm1P, m1, Nn, 64, 32, out_b1);
    }
    final_out<<<blocks(Nn, BT), BT, 0, stream>>>(m1, out_w2, out_b2, out, Nn);
}
